// AttentionalGraphAggregation_14353780703951
// MI455X (gfx1250) — matrix operand order repaired and checked by reading, not compiled, not run on hardware
//
#include <hip/hip_runtime.h>
#include <hip/hip_bf16.h>

typedef __attribute__((ext_vector_type(16))) __bf16 v16bf;
typedef __attribute__((ext_vector_type(8)))  float  v8f;

#define D_IN   128
#define D_HID  64
#define D_OUT  128
#define NSEGS  8192

// ---------- helpers ----------
__device__ __forceinline__ unsigned short f2bf_bits(float f) {
    unsigned u = __float_as_uint(f);
    unsigned r = u + 0x7FFFu + ((u >> 16) & 1u);   // RNE
    return (unsigned short)(r >> 16);
}
__device__ __forceinline__ unsigned fenc(float f) {   // order-preserving float->uint
    unsigned u = __float_as_uint(f);
    return (u >> 31) ? ~u : (u | 0x80000000u);
}
__device__ __forceinline__ float fdec(unsigned e) {
    unsigned u = (e >> 31) ? (e & 0x7FFFFFFFu) : ~e;
    return __uint_as_float(u);
}
__device__ __forceinline__ v16bf cvt16(const float* f) {
    v16bf r;
#pragma unroll
    for (int i = 0; i < 16; ++i) r[i] = (__bf16)f[i];
    return r;
}
// Build A fragment for 16x16x32 bf16 WMMA. rowp points at this lane's row of x.
// lanes 0-15: elems 0..15 = K kb..kb+15  (kb = ks*32)
// lanes16-31: elems 0..15 = K kb+16..kb+31
__device__ __forceinline__ v16bf load_a_frag(const float* rowp, int ks, int g) {
    alignas(16) float fa[16];
    int kb = ks * 32 + g * 16;
    *(float4*)(fa + 0)  = *(const float4*)(rowp + kb + 0);
    *(float4*)(fa + 4)  = *(const float4*)(rowp + kb + 4);
    *(float4*)(fa + 8)  = *(const float4*)(rowp + kb + 8);
    *(float4*)(fa + 12) = *(const float4*)(rowp + kb + 12);
    return cvt16(fa);
}

// ---------- kernel 0: init output + segment scratch ----------
__global__ void k_init(float* __restrict__ out, int out_n,
                       unsigned* __restrict__ segmax, float* __restrict__ denom, int nsegs) {
    int stride = gridDim.x * blockDim.x;
    for (int i = blockIdx.x * blockDim.x + threadIdx.x; i < out_n; i += stride) out[i] = 0.f;
    for (int i = blockIdx.x * blockDim.x + threadIdx.x; i < nsegs; i += stride) {
        segmax[i] = 0u;     // decodes to -NaN; only touched for non-empty segments
        denom[i]  = 0.f;
    }
}

// ---------- kernel 1: gate MLP (x@W1 relu @W2) + segment max ----------
__global__ __launch_bounds__(256) void k_gate(
    const float* __restrict__ x, const int* __restrict__ idx,
    const float* __restrict__ W1, const float* __restrict__ b1,
    const float* __restrict__ W2, const float* __restrict__ b2,
    float* __restrict__ gate, unsigned* __restrict__ segmax,
    int nrows, int ntiles)
{
    // W1^T packed bf16 in LDS: w1t[n*128 + k], so a lane's B fragment is 32 contiguous bytes
    __shared__ unsigned short w1t[D_HID * D_IN];
    for (int i = threadIdx.x; i < D_HID * D_IN; i += blockDim.x) {
        int n = i >> 7, k = i & 127;
        w1t[i] = f2bf_bits(W1[k * D_HID + n]);
    }
    __syncthreads();

    int wave = threadIdx.x >> 5;
    int lane = threadIdx.x & 31;
    int tile = blockIdx.x * (blockDim.x >> 5) + wave;
    if (tile >= ntiles) return;                       // whole-wave uniform: EXEC stays all-1s
    int l15 = lane & 15, g = lane >> 4;
    int row0 = tile << 4;
    const float* rowp = x + (long long)(row0 + l15) * D_IN;

    v8f acc[4];
#pragma unroll
    for (int nt = 0; nt < 4; ++nt)
#pragma unroll
        for (int i = 0; i < 8; ++i) acc[nt][i] = 0.f;

#pragma unroll
    for (int ks = 0; ks < 4; ++ks) {
        v16bf a = load_a_frag(rowp, ks, g);
        // preload all B fragments for this K-step (one ds clause, one wait),
        // then issue the 4 WMMAs back-to-back
        v16bf bfr[4];
#pragma unroll
        for (int nt = 0; nt < 4; ++nt)
            bfr[nt] = *reinterpret_cast<const v16bf*>(
                &w1t[(nt * 16 + l15) * D_IN + ks * 32 + g * 16]);
#pragma unroll
        for (int nt = 0; nt < 4; ++nt)
            acc[nt] = __builtin_amdgcn_wmma_f32_16x16x32_bf16(
                false, a, false, bfr[nt], (short)0, acc[nt], false, false);
    }

    // h = relu(acc + b1); gate = h @ W2 + b2 (per-row reduction over 64 cols)
    float b1v[4], w2v[4];
#pragma unroll
    for (int nt = 0; nt < 4; ++nt) {
        int c = nt * 16 + l15;
        b1v[nt] = b1[c];
        w2v[nt] = W2[c];       // W2 is [64,1]
    }
    float p[8];
#pragma unroll
    for (int r = 0; r < 8; ++r) {
        float s = 0.f;
#pragma unroll
        for (int nt = 0; nt < 4; ++nt)
            s += fmaxf(acc[nt][r] + b1v[nt], 0.f) * w2v[nt];
        p[r] = s;
    }
    // reduce across the 16 lanes of each D-matrix lane group
#pragma unroll
    for (int m = 1; m < 16; m <<= 1)
#pragma unroll
        for (int r = 0; r < 8; ++r) p[r] += __shfl_xor(p[r], m, 32);

    float b2v = b2[0];
    if (l15 == 0) {
#pragma unroll
        for (int r = 0; r < 8; ++r) {
            int row = row0 + g * 8 + r;               // D layout: VGPR r -> M = r + 8*group
            if (row < nrows) {
                float gv = p[r] + b2v;
                gate[row] = gv;
                atomicMax(&segmax[idx[row]], fenc(gv));
            }
        }
    }
}

// ---------- kernel 2: denom = segment_sum(exp(gate - max)); sorted-run merged atomics ----------
__global__ void k_denom(const float* __restrict__ gate, const int* __restrict__ idx,
                        const unsigned* __restrict__ segmax, float* __restrict__ denom, int nrows)
{
    long long t = (long long)blockIdx.x * blockDim.x + threadIdx.x;
    long long base = t * 16;
    if (base >= nrows) return;
    long long end = base + 16; if (end > nrows) end = nrows;
    int cs = -1; float acc = 0.f, m = 0.f;
    for (long long r = base; r < end; ++r) {
        int s = idx[r];
        if (s != cs) {
            if (cs >= 0) atomicAdd(&denom[cs], acc);
            cs = s; m = fdec(segmax[s]); acc = 0.f;
        }
        acc += __expf(gate[r] - m);
    }
    if (cs >= 0) atomicAdd(&denom[cs], acc);
}

// ---------- kernel 3: t = relu(x@Wt+bt); out[seg] += alpha * t (run-merged scatter) ----------
__global__ __launch_bounds__(256) void k_scatter(
    const float* __restrict__ x, const int* __restrict__ idx,
    const float* __restrict__ Wt, const float* __restrict__ bt,
    const float* __restrict__ gate, const unsigned* __restrict__ segmax,
    const float* __restrict__ denom, float* __restrict__ out,
    int nrows, int ntiles)
{
    __shared__ unsigned short wtT[D_OUT * D_IN];     // Wt^T packed bf16: wtT[n*128 + k]
    for (int i = threadIdx.x; i < D_OUT * D_IN; i += blockDim.x) {
        int n = i >> 7, k = i & 127;
        wtT[i] = f2bf_bits(Wt[k * D_OUT + n]);
    }
    __syncthreads();

    int wave = threadIdx.x >> 5;
    int lane = threadIdx.x & 31;
    int tile = blockIdx.x * (blockDim.x >> 5) + wave;
    if (tile >= ntiles) return;
    int l15 = lane & 15, g = lane >> 4;
    int row0 = tile << 4;
    const float* rowp = x + (long long)(row0 + l15) * D_IN;

    v8f acc[8];
#pragma unroll
    for (int nt = 0; nt < 8; ++nt)
#pragma unroll
        for (int i = 0; i < 8; ++i) acc[nt][i] = 0.f;

#pragma unroll
    for (int ks = 0; ks < 4; ++ks) {
        v16bf a = load_a_frag(rowp, ks, g);
        // preload this K-step's 8 B fragments (two clauses of 4), then 8 WMMAs back-to-back
        v16bf bfr[8];
#pragma unroll
        for (int nt = 0; nt < 8; ++nt)
            bfr[nt] = *reinterpret_cast<const v16bf*>(
                &wtT[(nt * 16 + l15) * D_IN + ks * 32 + g * 16]);
#pragma unroll
        for (int nt = 0; nt < 8; ++nt)
            acc[nt] = __builtin_amdgcn_wmma_f32_16x16x32_bf16(
                false, a, false, bfr[nt], (short)0, acc[nt], false, false);
    }

    float btv[8];
#pragma unroll
    for (int nt = 0; nt < 8; ++nt) btv[nt] = bt[nt * 16 + l15];

    // alpha per row of this tile handled by this lane group
    float alpha[8]; int segs[8];
#pragma unroll
    for (int r = 0; r < 8; ++r) {
        int row = row0 + g * 8 + r;
        int rc = row < nrows ? row : nrows - 1;
        int s = idx[rc];
        float m = fdec(segmax[s]);
        float d = denom[s];
        float a = __expf(gate[rc] - m) / (d + 1e-16f);
        alpha[r] = (row < nrows) ? a : 0.f;
        segs[r]  = s;
    }

    // scatter: merge consecutive rows sharing a segment (index is sorted) before atomics
#pragma unroll
    for (int nt = 0; nt < 8; ++nt) {
        int col = nt * 16 + l15;
        int cs = segs[0];
        float sum = fmaxf(acc[nt][0] + btv[nt], 0.f) * alpha[0];
#pragma unroll
        for (int r = 1; r < 8; ++r) {
            float v = fmaxf(acc[nt][r] + btv[nt], 0.f) * alpha[r];
            if (segs[r] == cs) { sum += v; }
            else {
                atomicAdd(out + (long long)cs * D_OUT + col, sum);
                cs = segs[r]; sum = v;
            }
        }
        atomicAdd(out + (long long)cs * D_OUT + col, sum);
    }
}

// ---------- host ----------
extern "C" void kernel_launch(void* const* d_in, const int* in_sizes, int n_in,
                              void* d_out, int out_size, void* d_ws, size_t ws_size,
                              hipStream_t stream) {
    const float* x  = (const float*)d_in[0];
    const int*   idx = (const int*) d_in[1];
    const float* W1 = (const float*)d_in[2];
    const float* b1 = (const float*)d_in[3];
    const float* W2 = (const float*)d_in[4];
    const float* b2 = (const float*)d_in[5];
    const float* Wt = (const float*)d_in[6];
    const float* bt = (const float*)d_in[7];

    int nrows = in_sizes[0] / D_IN;

    float*    gate   = (float*)d_ws;                 // nrows floats
    unsigned* segmax = (unsigned*)(gate + nrows);    // NSEGS uints
    float*    denom  = (float*)(segmax + NSEGS);     // NSEGS floats

    k_init<<<4096, 256, 0, stream>>>((float*)d_out, out_size, segmax, denom, NSEGS);

    int ntiles = (nrows + 15) >> 4;
    int blocks = (ntiles + 7) >> 3;                  // 8 waves per block, 1 tile per wave
    k_gate<<<blocks, 256, 0, stream>>>(x, idx, W1, b1, W2, b2, gate, segmax, nrows, ntiles);

    int t2 = (nrows + 15) >> 4;
    k_denom<<<(t2 + 255) / 256, 256, 0, stream>>>(gate, idx, segmax, denom, nrows);

    k_scatter<<<blocks, 256, 0, stream>>>(x, idx, Wt, bt, gate, segmax, denom,
                                          (float*)d_out, nrows, ntiles);
}
